// StrategyAwareStringScalarOps_44341242364597
// MI455X (gfx1250) — matrix operand order repaired and checked by reading, not compiled, not run on hardware
//
#include <hip/hip_runtime.h>

#define NSTR 6
#define NSCL 20
#define DM   256
#define TOPK 4
#define LN_EPS 1e-5f

typedef _Float16 h16 __attribute__((ext_vector_type(16)));
typedef _Float16 h8  __attribute__((ext_vector_type(8)));
typedef float    f8  __attribute__((ext_vector_type(8)));

union H16u { h16 v; h8 h[2]; };

// weight fragment offsets inside d_ws (in halves, after 64-byte meta header)
constexpr int OFF_CM1 = 0;            // 768x256
constexpr int OFF_CM2 = 196608;       // 256x256
constexpr int OFF_DF1 = 262144;       // 512x256
constexpr int OFF_DF2 = 393216;       // 256x256
constexpr int OFF_FU1 = 458752;       // 768x512
constexpr int OFF_FU2 = 851968;       // 512x256  (end = 983040 halves = 1.92MB)

__device__ __forceinline__ h8 ld8(const _Float16* p){ return *(const h8*)p; }
__device__ __forceinline__ h16 mk16(h8 lo, h8 hi){ H16u u; u.h[0]=lo; u.h[1]=hi; return u.v; }
__device__ __forceinline__ h8 habs8(h8 x){
  union { h8 h; unsigned short u[8]; } a; a.h = x;
  #pragma unroll
  for (int i=0;i<8;++i) a.u[i] &= 0x7FFFu;
  return a.h;
}
__device__ __forceinline__ f8 wmma_f16(h16 a, h16 b, f8 c){
  // D = A(16x32 f16) x B(32x16 f16) + C(16x16 f32)
  return __builtin_amdgcn_wmma_f32_16x16x32_f16(false, a, false, b, (short)0, c, false, false);
}

// ---------------- weight prep: f32 (K x N) row-major -> f16 B-fragment layout ----------------
// layout: idx = ((nt*(K/32) + kb)*32 + lane)*16 + h ; element = W[kb*32 + 16*(h>>3) + 8*(lane>>4) + (h&7)][nt*16 + (lane&15)]
__global__ void prep_weights(const float* __restrict__ src, _Float16* __restrict__ dst, int K, int N){
  int idx = blockIdx.x*blockDim.x + threadIdx.x;
  if (idx >= K*N) return;
  int h    = idx & 15;
  int t    = idx >> 4;
  int lane = t & 31;
  int t2   = t >> 5;
  int KB   = K >> 5;
  int kb   = t2 % KB;
  int nt   = t2 / KB;
  int k    = kb*32 + ((h >> 3) << 4) + ((lane >> 4) << 3) + (h & 7);
  int nc   = nt*16 + (lane & 15);
  dst[idx] = (_Float16)src[k*N + nc];
}

// ---------------- top-k(4) of 6x20 scores + softmax weights ----------------
__global__ void topk_kernel(const float* __restrict__ ps, const float* __restrict__ temp_p,
                            float* __restrict__ meta_w, int* __restrict__ meta_ti, int* __restrict__ meta_tj){
  if (threadIdx.x != 0 || blockIdx.x != 0) return;
  float v[NSTR*NSCL];
  bool used[NSTR*NSCL];
  for (int i=0;i<NSTR*NSCL;++i){ v[i]=ps[i]; used[i]=false; }
  float tv[TOPK]; int tix[TOPK];
  for (int s=0;s<TOPK;++s){
    float best=-3.402823466e38f; int bi=0;
    for (int i=0;i<NSTR*NSCL;++i) if (!used[i] && v[i] > best){ best=v[i]; bi=i; }
    used[bi]=true; tv[s]=best; tix[s]=bi;
  }
  float temp = fmaxf(temp_p[0], 0.1f);
  float mx = tv[0], e[TOPK], se=0.f;
  for (int s=0;s<TOPK;++s){ e[s]=expf((tv[s]-mx)/temp); se+=e[s]; }
  for (int s=0;s<TOPK;++s){
    meta_w[s]=e[s]/se; meta_ti[s]=tix[s]/NSCL; meta_tj[s]=tix[s]%NSCL;
  }
}

// ---------------- fused  relu(X @ W1 + b1) @ W2 + b2  with register accumulators ----------------
// X is 16 x (in_kb*32), fed via aload(kb) -> A fragment.  Output is 16x256 in acc2 (16 tiles of v8f).
// Hidden dim processed in groups of 64 columns: one A-fragment load feeds 4 independent WMMAs.
// B-fragments are batch-loaded (4 at a time) ahead of the WMMA group so the loads form one
// clause with a single counter wait, instead of a serialized load->wait->wmma chain.
template<typename AF>
__device__ __forceinline__ void fused_pair(AF&& aload, int in_kb,
    const _Float16* __restrict__ w1f, const float* __restrict__ b1, int hid_w,
    const _Float16* __restrict__ w2f, const float* __restrict__ b2,
    f8 acc2[16], _Float16* __restrict__ Ht /*16x64 f16*/, int lane)
{
  const int n = lane & 15, hi = lane >> 4, m = n;
  const h16* W1 = (const h16*)w1f;
  const h16* W2 = (const h16*)w2f;
  #pragma unroll
  for (int nt=0; nt<16; ++nt){
    float bv = b2[nt*16 + n];
    f8 t;
    #pragma unroll
    for (int r=0;r<8;++r) t[r]=bv;
    acc2[nt]=t;
  }
  const int HPB = hid_w >> 5;           // hidden 32-chunks (W2 K-block count)
  const int HGB = hid_w >> 6;           // hidden 64-groups
  for (int hg=0; hg<HGB; ++hg){
    f8 a1[4];
    #pragma unroll
    for (int j=0;j<4;++j){
      float b1v = b1[(hg*4 + j)*16 + n];
      #pragma unroll
      for (int r=0;r<8;++r) a1[j][r]=b1v;
    }
    for (int kb=0; kb<in_kb; ++kb){
      h16 a = aload(kb);                // loaded/computed ONCE per 4 hidden chunks
      h16 bf[4];
      #pragma unroll
      for (int j=0;j<4;++j) bf[j] = W1[((hg*4 + j)*in_kb + kb)*32 + lane];
      #pragma unroll
      for (int j=0;j<4;++j) a1[j] = wmma_f16(a, bf[j], a1[j]);
    }
    #pragma unroll
    for (int j=0;j<4;++j)
      #pragma unroll
      for (int r=0;r<8;++r)
        Ht[(r + 8*hi)*64 + j*16 + n] = (_Float16)fmaxf(a1[j][r], 0.f);
    __syncthreads();
    #pragma unroll
    for (int hp2=0; hp2<2; ++hp2){
      h16 ah = mk16(ld8(&Ht[m*64 + hp2*32 + 8*hi]), ld8(&Ht[m*64 + hp2*32 + 16 + 8*hi]));
      #pragma unroll
      for (int ng=0; ng<4; ++ng){
        h16 bf[4];
        #pragma unroll
        for (int q=0;q<4;++q) bf[q] = W2[((ng*4 + q)*HPB + hg*2 + hp2)*32 + lane];
        #pragma unroll
        for (int q=0;q<4;++q) acc2[ng*4 + q] = wmma_f16(ah, bf[q], acc2[ng*4 + q]);
      }
    }
    __syncthreads();
  }
}

// ---------------- main kernel: one wave handles 16 batch rows x all 4 top-k pairs ----------------
__global__ __launch_bounds__(32) void strategy_main_kernel(
  const float* __restrict__ strr, const float* __restrict__ scl,
  const float* __restrict__ cm_b1, const float* __restrict__ cm_b2,
  const float* __restrict__ df_b1, const float* __restrict__ df_b2,
  const float* __restrict__ fu_b1, const float* __restrict__ fu_b2,
  const float* __restrict__ ln_g,  const float* __restrict__ ln_b,
  const _Float16* __restrict__ wf,
  const float* __restrict__ meta_w, const int* __restrict__ meta_ti, const int* __restrict__ meta_tj,
  float* __restrict__ out, int B)
{
  __shared__ _Float16 sS [16*DM];   // 8 KB  string rows (f16)
  __shared__ _Float16 sC [16*DM];   // 8 KB  scalar rows (f16)
  __shared__ _Float16 sCF[16*DM];   // 8 KB  cross_f (f16)
  __shared__ _Float16 sDF[16*DM];   // 8 KB  diff_f  (f16)
  __shared__ _Float16 sHt[16*64];   // 2 KB  hidden tile (64 cols)
  __shared__ float    sOut[16*DM];  // 16 KB weighted-sum accumulator

  const int lane = threadIdx.x;
  const int n = lane & 15, hi = lane >> 4, m = n;
  const int b0 = blockIdx.x * 16;

  for (int e = lane; e < 16*DM; e += 32) sOut[e] = 0.f;

  f8 acc2[16];

  for (int k=0; k<TOPK; ++k){
    const int   tik = meta_ti[k];
    const int   tjk = meta_tj[k];
    const float wk  = meta_w[k];
    __syncthreads();
    // stage s, c rows -> LDS f16
    for (int e4 = lane; e4 < (16*DM)/4; e4 += 32){
      int e = e4*4, row = e >> 8, col = e & 255;
      int gb = b0 + row; if (gb >= B) gb = B-1;
      const float4 vs = *(const float4*)(strr + (((size_t)gb*NSTR + tik) << 8) + col);
      const float4 vc = *(const float4*)(scl  + (((size_t)gb*NSCL + tjk) << 8) + col);
      sS[e+0]=(_Float16)vs.x; sS[e+1]=(_Float16)vs.y; sS[e+2]=(_Float16)vs.z; sS[e+3]=(_Float16)vs.w;
      sC[e+0]=(_Float16)vc.x; sC[e+1]=(_Float16)vc.y; sC[e+2]=(_Float16)vc.z; sC[e+3]=(_Float16)vc.w;
    }
    __syncthreads();

    // ---- cross MLP: [s, c, s*c] (768) -> 256 -> 256 ----
    auto a_cm = [&](int kb)->h16{
      if (kb < 8){  int c0 = kb*32 + 8*hi;      return mk16(ld8(&sS[m*DM+c0]), ld8(&sS[m*DM+c0+16])); }
      if (kb < 16){ int c0 = (kb-8)*32 + 8*hi;  return mk16(ld8(&sC[m*DM+c0]), ld8(&sC[m*DM+c0+16])); }
      int c0 = (kb-16)*32 + 8*hi;
      return mk16(ld8(&sS[m*DM+c0])    * ld8(&sC[m*DM+c0]),
                  ld8(&sS[m*DM+c0+16]) * ld8(&sC[m*DM+c0+16]));
    };
    fused_pair(a_cm, 24, wf+OFF_CM1, cm_b1, 256, wf+OFF_CM2, cm_b2, acc2, sHt, lane);
    #pragma unroll
    for (int nt=0; nt<16; ++nt)
      #pragma unroll
      for (int r=0; r<8; ++r)
        sCF[(r+8*hi)*DM + nt*16 + n] = (_Float16)acc2[nt][r];
    __syncthreads();

    // ---- diff MLP: [d, |d|] (512) -> 256 -> 256 ----
    auto a_df = [&](int kb)->h16{
      int c0 = (kb & 7)*32 + 8*hi;
      h8 d0 = ld8(&sS[m*DM+c0])    - ld8(&sC[m*DM+c0]);
      h8 d1 = ld8(&sS[m*DM+c0+16]) - ld8(&sC[m*DM+c0+16]);
      if (kb < 8) return mk16(d0, d1);
      return mk16(habs8(d0), habs8(d1));
    };
    fused_pair(a_df, 16, wf+OFF_DF1, df_b1, 256, wf+OFF_DF2, df_b2, acc2, sHt, lane);
    #pragma unroll
    for (int nt=0; nt<16; ++nt)
      #pragma unroll
      for (int r=0; r<8; ++r)
        sDF[(r+8*hi)*DM + nt*16 + n] = (_Float16)acc2[nt][r];
    __syncthreads();

    // ---- fuse MLP: [cross_f, diff_f, s*c] (768) -> 512 -> 256 ----
    auto a_fu = [&](int kb)->h16{
      if (kb < 8){  int c0 = kb*32 + 8*hi;      return mk16(ld8(&sCF[m*DM+c0]), ld8(&sCF[m*DM+c0+16])); }
      if (kb < 16){ int c0 = (kb-8)*32 + 8*hi;  return mk16(ld8(&sDF[m*DM+c0]), ld8(&sDF[m*DM+c0+16])); }
      int c0 = (kb-16)*32 + 8*hi;
      return mk16(ld8(&sS[m*DM+c0])    * ld8(&sC[m*DM+c0]),
                  ld8(&sS[m*DM+c0+16]) * ld8(&sC[m*DM+c0+16]));
    };
    fused_pair(a_fu, 24, wf+OFF_FU1, fu_b1, 512, wf+OFF_FU2, fu_b2, acc2, sHt, lane);

    // ---- layernorm over each row (M = r + 8*hi), half-wave butterfly reduction ----
    float S[8], Q[8];
    #pragma unroll
    for (int r=0;r<8;++r){ S[r]=0.f; Q[r]=0.f; }
    #pragma unroll
    for (int nt=0; nt<16; ++nt)
      #pragma unroll
      for (int r=0; r<8; ++r){ float v = acc2[nt][r]; S[r]+=v; Q[r]+=v*v; }
    #pragma unroll
    for (int r=0;r<8;++r){
      #pragma unroll
      for (int mask=1; mask<16; mask<<=1){
        S[r] += __shfl_xor(S[r], mask, 32);
        Q[r] += __shfl_xor(Q[r], mask, 32);
      }
    }
    float mu[8], rs[8];
    #pragma unroll
    for (int r=0;r<8;++r){
      mu[r] = S[r] * (1.f/256.f);
      float var = Q[r]*(1.f/256.f) - mu[r]*mu[r];
      rs[r] = rsqrtf(var + LN_EPS);
    }
    #pragma unroll
    for (int nt=0; nt<16; ++nt){
      float g = ln_g[nt*16+n], bb = ln_b[nt*16+n];
      #pragma unroll
      for (int r=0; r<8; ++r){
        float val = (acc2[nt][r]-mu[r])*rs[r]*g + bb;
        sOut[(r+8*hi)*DM + nt*16 + n] += wk*val;
      }
    }
    __syncthreads();
  }

  // coalesced float4 stores of the weighted sum
  for (int e4 = lane; e4 < (16*DM)/4; e4 += 32){
    int e = e4*4, row = e >> 8, col = e & 255;
    int gb = b0 + row;
    if (gb < B){
      float4 v;
      v.x = sOut[e+0]; v.y = sOut[e+1]; v.z = sOut[e+2]; v.w = sOut[e+3];
      *(float4*)(out + ((size_t)gb << 8) + col) = v;
    }
  }
}

extern "C" void kernel_launch(void* const* d_in, const int* in_sizes, int n_in,
                              void* d_out, int out_size, void* d_ws, size_t ws_size,
                              hipStream_t stream){
  const float* strr  = (const float*)d_in[0];
  const float* scl   = (const float*)d_in[1];
  const float* ps    = (const float*)d_in[2];
  const float* temp  = (const float*)d_in[3];
  const float* cm_w1 = (const float*)d_in[4];  const float* cm_b1 = (const float*)d_in[5];
  const float* cm_w2 = (const float*)d_in[6];  const float* cm_b2 = (const float*)d_in[7];
  const float* df_w1 = (const float*)d_in[8];  const float* df_b1 = (const float*)d_in[9];
  const float* df_w2 = (const float*)d_in[10]; const float* df_b2 = (const float*)d_in[11];
  const float* fu_w1 = (const float*)d_in[12]; const float* fu_b1 = (const float*)d_in[13];
  const float* fu_w2 = (const float*)d_in[14]; const float* fu_b2 = (const float*)d_in[15];
  const float* ln_g  = (const float*)d_in[16]; const float* ln_b  = (const float*)d_in[17];
  float* out = (float*)d_out;
  const int B = in_sizes[0] / (NSTR * DM);

  // d_ws layout: [0,16) meta_w floats, [16,32) meta_ti ints, [32,48) meta_tj ints, [64, ...) f16 weight frags (~1.92MB)
  float*    meta_w  = (float*)d_ws;
  int*      meta_ti = (int*)d_ws + 4;
  int*      meta_tj = (int*)d_ws + 8;
  _Float16* wf      = (_Float16*)((char*)d_ws + 64);

  topk_kernel<<<1, 1, 0, stream>>>(ps, temp, meta_w, meta_ti, meta_tj);

  auto prep = [&](const float* src, _Float16* dst, int K, int N){
    int tot = K * N;
    prep_weights<<<(tot + 255) / 256, 256, 0, stream>>>(src, dst, K, N);
  };
  prep(cm_w1, wf + OFF_CM1, 768, 256);
  prep(cm_w2, wf + OFF_CM2, 256, 256);
  prep(df_w1, wf + OFF_DF1, 512, 256);
  prep(df_w2, wf + OFF_DF2, 256, 256);
  prep(fu_w1, wf + OFF_FU1, 768, 512);
  prep(fu_w2, wf + OFF_FU2, 512, 256);

  strategy_main_kernel<<<dim3((B + 15) / 16), 32, 0, stream>>>(
      strr, scl, cm_b1, cm_b2, df_b1, df_b2, fu_b1, fu_b2, ln_g, ln_b,
      wf, meta_w, meta_ti, meta_tj, out, B);
}
